// SwinBlock_36893769072673
// MI455X (gfx1250) — compile-verified, matrix operand order checked by reading, not run on hardware
//
#include <hip/hip_runtime.h>
#include <hip/hip_bf16.h>
#include <math.h>

// ---------------------------------------------------------------------------
// Swin block forward for gfx1250 (MI455X).  wave32, WMMA f32_16x16x32_f16,
// TDM (tensor_load_to_lds) for GEMM A-tile staging.
//
// WMMA operand layout assumptions (CDNA5 ISA 7.12.2):
//   A (16x32 f16): lane = {half,row}: row = lane&15, half = lane>>4.
//     element e of v16h:  k = (e<8) ? half*8+e : 16+half*8+(e-8)
//     -> each fragment = two contiguous 16-byte chunks per lane.
//   B (32x16 f16): mirror of A with lane&15 = output column n, same k packing
//     (i.e. B supplied as B^T rows per lane).  C/D: vgpr g, lanes0-15 m=g,
//     lanes16-31 m=g+8, n=lane&15.
// ---------------------------------------------------------------------------

typedef __attribute__((ext_vector_type(16))) _Float16 v16h;
typedef __attribute__((ext_vector_type(8)))  _Float16 v8h;
typedef __attribute__((ext_vector_type(8)))  float    v8f;
typedef __attribute__((ext_vector_type(4)))  unsigned int v4u;
typedef __attribute__((ext_vector_type(8)))  int v8i;
typedef __attribute__((ext_vector_type(4)))  int v4i;

#define RES_SCALE 0.70710678118654752f
#define NTOK 32768      // 8 * 64 * 64
#define CDIM 384
#define NHEADS 12
#define HDIM 32
#define SHIFT_AMT 4

__device__ __forceinline__ v16h frag_ld(const _Float16* p) {
  v8h lo = *(const v8h*)p;
  v8h hi = *(const v8h*)(p + 16);
  return __builtin_shufflevector(lo, hi, 0,1,2,3,4,5,6,7,8,9,10,11,12,13,14,15);
}

__device__ __forceinline__ v8f wmma16(v16h a, v16h b, v8f c) {
  return __builtin_amdgcn_wmma_f32_16x16x32_f16(false, a, false, b, (short)0, c,
                                                false, false);
}

__device__ __forceinline__ int kpack(int e, int half) {
  return (e < 8) ? (half * 8 + e) : (16 + half * 8 + (e - 8));
}

// ---------------------------------------------------------------------------
// TDM: 2D tile load (f16 elements) Global -> LDS, one instruction per tile.
// D# layout per CDNA5 ISA 8.3/8.4:
//   group0: [1:0]=count=1, [63:32]=lds_addr, [120:64]=global_addr,
//           [127:126]=type=2
//   group1: [17:16]=data_size(1=2B), [79:48]=tensor_dim0, [111:80]=tensor_dim1,
//           [127:112]=tile_dim0, [143:128]=tile_dim1,
//           [207:160]=tensor_dim0_stride (data_size units)
// groups 2/3 zero (<=2D tensor).  Issued by one wave; tracked by TENSORcnt.
// Toolchain note: this build uses the 6-arg builtin form
//   (v4u, v8i, v4i, v4i, v8i, i32 cpol) -- extra group zero-filled.
// ---------------------------------------------------------------------------
__device__ __forceinline__ void tdm_load_2d_f16(unsigned lds_off,
                                                const _Float16* gptr,
                                                unsigned tile_k,
                                                unsigned tile_rows,
                                                unsigned long long stride_elems) {
  unsigned long long ga = (unsigned long long)(size_t)(const void*)gptr;
  v4u g0;
  g0[0] = 1u;                                   // count=1, user descriptor
  g0[1] = lds_off;                              // LDS byte address
  g0[2] = (unsigned)(ga & 0xFFFFFFFFu);         // global_addr[31:0]
  g0[3] = (unsigned)((ga >> 32) & 0x01FFFFFFu)  // global_addr[56:32]
        | (2u << 30);                           // type = 2 ("image")
  v8i g1;
  g1[0] = (int)(1u << 16);                                       // data_size=2B
  g1[1] = (int)((tile_k & 0xFFFFu) << 16);                       // tdim0 lo
  g1[2] = (int)(((tile_k >> 16) & 0xFFFFu) |
                ((tile_rows & 0xFFFFu) << 16));                  // tdim0 hi|tdim1 lo
  g1[3] = (int)(((tile_rows >> 16) & 0xFFFFu) |
                ((tile_k & 0xFFFFu) << 16));                     // tdim1 hi|tile_dim0
  g1[4] = (int)(tile_rows & 0xFFFFu);                            // tile_dim1 (dim2=0)
  g1[5] = (int)(unsigned)(stride_elems & 0xFFFFFFFFull);         // stride0[31:0]
  g1[6] = (int)(unsigned)((stride_elems >> 32) & 0xFFFFull);     // stride0[47:32]
  g1[7] = 0;
  v4i gz4 = (v4i){0, 0, 0, 0};
  v8i gz8 = (v8i){0, 0, 0, 0, 0, 0, 0, 0};
  __builtin_amdgcn_tensor_load_to_lds(g0, g1, gz4, gz4, gz8, 0);
}

// ---------------------------------------------------------------------------
// K0: mod tables  mod{1,2}[b, 0:768] = t_emb[b] @ tp{1,2}_w + tp{1,2}_b
// ---------------------------------------------------------------------------
__global__ void time_mod_kernel(const float* __restrict__ t_emb,
                                const float* __restrict__ w1, const float* __restrict__ b1,
                                const float* __restrict__ w2, const float* __restrict__ b2,
                                float* __restrict__ mod1, float* __restrict__ mod2) {
  int i = blockIdx.x * blockDim.x + threadIdx.x;
  if (i >= 2 * 8 * 768) return;
  int which = i / (8 * 768);
  int rem   = i % (8 * 768);
  int b     = rem / 768;
  int col   = rem % 768;
  const float* w  = which ? w2 : w1;
  const float* bb = which ? b2 : b1;
  float acc = bb[col];
  for (int k = 0; k < 384; ++k)
    acc += t_emb[b * 384 + k] * w[k * 768 + col];
  (which ? mod2 : mod1)[b * 768 + col] = acc;
}

// ---------------------------------------------------------------------------
// K1: f32 -> f16 weight conversion
// ---------------------------------------------------------------------------
__global__ void cvt_f16_kernel(const float* __restrict__ src,
                               _Float16* __restrict__ dst, int n) {
  int i = blockIdx.x * blockDim.x + threadIdx.x;
  if (i < n) dst[i] = (_Float16)src[i];
}

// ---------------------------------------------------------------------------
// K2: modulated RMSNorm + cyclic shift + window partition.
// One wave per windowed token; lane-strided channels; xor-shuffle reduction.
// windowed coord r  <->  original coord (r + SHIFT) & 63
// ---------------------------------------------------------------------------
__global__ __launch_bounds__(256) void norm1_window_kernel(
    const float* __restrict__ x, const float* __restrict__ mod1,
    const float* __restrict__ n1w, _Float16* __restrict__ Xwin) {
  int wave = threadIdx.x >> 5, lane = threadIdx.x & 31;
  int token = blockIdx.x * 8 + wave;            // windowed order: win*64 + t
  int win = token >> 6, tt = token & 63;
  int b = win >> 6, wy = (win >> 3) & 7, wx = win & 7;
  int y  = ((wy * 8 + (tt >> 3)) + SHIFT_AMT) & 63;
  int xq = ((wx * 8 + (tt & 7))  + SHIFT_AMT) & 63;
  float v[12];
  float ss = 0.f;
#pragma unroll
  for (int j = 0; j < 12; ++j) {
    int c = lane + 32 * j;
    v[j] = x[((size_t)(b * CDIM + c)) * 4096 + y * 64 + xq];
    ss += v[j] * v[j];
  }
#pragma unroll
  for (int off = 1; off < 32; off <<= 1) ss += __shfl_xor(ss, off, 32);
  float rs = rsqrtf(ss * (1.f / CDIM) + 1e-6f);
  const float* sc = mod1 + b * 768;
  const float* sh = sc + 384;
#pragma unroll
  for (int j = 0; j < 12; ++j) {
    int c = lane + 32 * j;
    float o = v[j] * rs * n1w[c] * (1.f + sc[c]) + sh[c];
    Xwin[(size_t)token * CDIM + c] = (_Float16)o;
  }
}

// ---------------------------------------------------------------------------
// K6: modulated RMSNorm (linear token order, rows of H1 are contiguous)
// ---------------------------------------------------------------------------
__global__ __launch_bounds__(256) void norm2_kernel(
    const float* __restrict__ H1, const float* __restrict__ mod2,
    const float* __restrict__ n2w, _Float16* __restrict__ X2) {
  int wave = threadIdx.x >> 5, lane = threadIdx.x & 31;
  int token = blockIdx.x * 8 + wave;
  int b = token >> 12;
  float v[12];
  float ss = 0.f;
#pragma unroll
  for (int j = 0; j < 12; ++j) {
    int c = lane + 32 * j;
    v[j] = H1[(size_t)token * CDIM + c];
    ss += v[j] * v[j];
  }
#pragma unroll
  for (int off = 1; off < 32; off <<= 1) ss += __shfl_xor(ss, off, 32);
  float rs = rsqrtf(ss * (1.f / CDIM) + 1e-6f);
  const float* sc = mod2 + b * 768;
  const float* sh = sc + 384;
#pragma unroll
  for (int j = 0; j < 12; ++j) {
    int c = lane + 32 * j;
    float o = v[j] * rs * n2w[c] * (1.f + sc[c]) + sh[c];
    X2[(size_t)token * CDIM + c] = (_Float16)o;
  }
}

// ---------------------------------------------------------------------------
// WMMA GEMM, C = A[MxK] @ B[KxN], f16 in, f32 accumulate.
// Block = 256 threads (8 waves), tile 128(M) x 64(N), BK = 64.
// A tile (128x64, row-major, stride K) staged by TDM from wave 0 (one
// tensor_load_to_lds per iteration); B tile staged manually transposed.
// Wave (wm 0..3, wn 0..1) computes 32x32 = 2x2 WMMA tiles, 8 WMMA/iter.
// Epilogues:
//   0: store f16 (qkv, no bias)
//   1: proj: +bias, un-shift window, +residual(x NCHW), *RES_SCALE -> H1 (NHWC f32)
//   2: +bias, store f16 (fc1 raw)
//   3: fc2: +bias, +H1 residual, *RES_SCALE, NHWC->NCHW store to out (f32)
// ---------------------------------------------------------------------------
template <int EPI>
__global__ __launch_bounds__(256) void gemm_kernel(
    const _Float16* __restrict__ A, const _Float16* __restrict__ Bw,
    int M, int N, int K,
    _Float16* __restrict__ outH, const float* __restrict__ bias,
    const float* __restrict__ xin, float* __restrict__ H1,
    float* __restrict__ outF) {
  __shared__ __align__(16) _Float16 As[128 * 64];
  __shared__ __align__(16) _Float16 Bs[64 * 64];   // transposed: Bs[n*64 + k]

  const int t    = threadIdx.x;
  const int lane = t & 31;
  const int wave = t >> 5;
  const int wm   = wave >> 1;
  const int wn   = wave & 1;
  const int r    = lane & 15;
  const int hh   = lane >> 4;

  const int m0 = blockIdx.y * 128;
  const int n0 = blockIdx.x * 64;

  const unsigned as_lds = (unsigned)(size_t)(void*)As;

  v8f acc[2][2];
#pragma unroll
  for (int i = 0; i < 2; ++i)
#pragma unroll
    for (int j = 0; j < 2; ++j) acc[i][j] = (v8f){};

  for (int kb = 0; kb < K; kb += 64) {
    // A tile 128x64 via Tensor Data Mover (wave 0 issues; TENSORcnt tracked).
    if (wave == 0)
      tdm_load_2d_f16(as_lds, A + (size_t)m0 * K + kb, 64u, 128u,
                      (unsigned long long)K);
    // B tile 64x64, stored transposed into Bs[n*64+k].  16 f16 per thread.
#pragma unroll
    for (int c = 0; c < 2; ++c) {
      int idx = t * 16 + c * 8;
      int bk  = idx >> 6;
      int nc  = idx & 63;
      v8h bv = *(const v8h*)(Bw + (size_t)(kb + bk) * N + n0 + nc);
#pragma unroll
      for (int i = 0; i < 8; ++i) Bs[(nc + i) * 64 + bk] = bv[i];
    }
    if (wave == 0) __builtin_amdgcn_s_wait_tensorcnt(0);
    __syncthreads();

    v16h af[2][2], bf[2][2];
#pragma unroll
    for (int i = 0; i < 2; ++i)
#pragma unroll
      for (int kc = 0; kc < 2; ++kc)
        af[i][kc] = frag_ld(As + (wm * 32 + i * 16 + r) * 64 + kc * 32 + hh * 8);
#pragma unroll
    for (int j = 0; j < 2; ++j)
#pragma unroll
      for (int kc = 0; kc < 2; ++kc)
        bf[j][kc] = frag_ld(Bs + (wn * 32 + j * 16 + r) * 64 + kc * 32 + hh * 8);
#pragma unroll
    for (int i = 0; i < 2; ++i)
#pragma unroll
      for (int j = 0; j < 2; ++j)
#pragma unroll
        for (int kc = 0; kc < 2; ++kc)
          acc[i][j] = wmma16(af[i][kc], bf[j][kc], acc[i][j]);
    __syncthreads();
  }

  // Epilogue.  C/D layout: m_local = hh*8 + g, n_local = r.
#pragma unroll
  for (int i = 0; i < 2; ++i) {
#pragma unroll
    for (int j = 0; j < 2; ++j) {
#pragma unroll
      for (int g = 0; g < 8; ++g) {
        int gm = m0 + wm * 32 + i * 16 + hh * 8 + g;
        int gn = n0 + wn * 32 + j * 16 + r;
        float d = acc[i][j][g];
        if (EPI == 0) {
          outH[(size_t)gm * N + gn] = (_Float16)d;
        } else if (EPI == 1) {
          d += bias[gn];
          int win = gm >> 6, tt = gm & 63;
          int b = win >> 6, wy = (win >> 3) & 7, wx = win & 7;
          int y  = ((wy * 8 + (tt >> 3)) + SHIFT_AMT) & 63;
          int xq = ((wx * 8 + (tt & 7))  + SHIFT_AMT) & 63;
          float resid = xin[((size_t)(b * CDIM + gn)) * 4096 + y * 64 + xq];
          H1[((size_t)(b * 4096 + y * 64 + xq)) * CDIM + gn] =
              (resid + d) * RES_SCALE;
        } else if (EPI == 2) {
          outH[(size_t)gm * N + gn] = (_Float16)(d + bias[gn]);
        } else {  // EPI == 3
          d += bias[gn];
          float o = (H1[(size_t)gm * CDIM + gn] + d) * RES_SCALE;
          int b = gm >> 12, pix = gm & 4095;
          outF[((size_t)(b * CDIM + gn)) * 4096 + pix] = o;
        }
      }
    }
  }
}

// ---------------------------------------------------------------------------
// Window attention.  One wave per (window, head); 4 waves / block.
// q,k RMS-normed in fragment registers; S = qk^T (16 WMMA); inline rel-bias;
// softmax via half-wave xor reductions; P staged through LDS; O = P@V (16 WMMA).
// ---------------------------------------------------------------------------
__global__ __launch_bounds__(128) void attn_kernel(
    const _Float16* __restrict__ qkv, const float* __restrict__ qn,
    const float* __restrict__ kn, const float* __restrict__ relb,
    _Float16* __restrict__ attn_out) {
  __shared__ __align__(16) _Float16 P[4][64 * 64];

  const int wave = threadIdx.x >> 5;
  const int lane = threadIdx.x & 31;
  const int gw   = blockIdx.x * 4 + wave;   // 0..6143
  const int win  = gw / NHEADS;
  const int head = gw % NHEADS;
  const int r  = lane & 15;
  const int hh = lane >> 4;

  const _Float16* rowbase = qkv + (size_t)win * 64 * (3 * CDIM) + head * HDIM;

  // Per-lane norm-weight vectors (d depends only on lane half).
  float wq[16], wk[16];
#pragma unroll
  for (int e = 0; e < 16; ++e) {
    int d = kpack(e, hh);
    wq[e] = qn[d] * 0.17677669529663687f;   // * HD^-0.5 folded into q
    wk[e] = kn[d];
  }

  // Q fragments (A operand), RMS-normed per row.
  v16h qf[4];
#pragma unroll
  for (int ti = 0; ti < 4; ++ti) {
    const _Float16* p = rowbase + (size_t)(ti * 16 + r) * (3 * CDIM) + hh * 8;
    v8h lo = *(const v8h*)p;
    v8h hi = *(const v8h*)(p + 16);
    float ss = 0.f;
#pragma unroll
    for (int e = 0; e < 8; ++e)
      ss += (float)lo[e] * (float)lo[e] + (float)hi[e] * (float)hi[e];
    ss += __shfl_xor(ss, 16, 32);
    float rs = rsqrtf(ss * (1.f / HDIM) + 1e-6f);
#pragma unroll
    for (int e = 0; e < 16; ++e) {
      float val = (e < 8) ? (float)lo[e] : (float)hi[e - 8];
      qf[ti][e] = (_Float16)(val * rs * wq[e]);
    }
  }

  // K fragments (B operand: lane column = key token), RMS-normed per key.
  v16h kf[4];
#pragma unroll
  for (int tj = 0; tj < 4; ++tj) {
    const _Float16* p = rowbase + (size_t)(tj * 16 + r) * (3 * CDIM) + CDIM + hh * 8;
    v8h lo = *(const v8h*)p;
    v8h hi = *(const v8h*)(p + 16);
    float ss = 0.f;
#pragma unroll
    for (int e = 0; e < 8; ++e)
      ss += (float)lo[e] * (float)lo[e] + (float)hi[e] * (float)hi[e];
    ss += __shfl_xor(ss, 16, 32);
    float rs = rsqrtf(ss * (1.f / HDIM) + 1e-6f);
#pragma unroll
    for (int e = 0; e < 16; ++e) {
      float val = (e < 8) ? (float)lo[e] : (float)hi[e - 8];
      kf[tj][e] = (_Float16)(val * rs * wk[e]);
    }
  }

  // S = q @ k^T : 4x4 tiles, single K-step (HD = 32).
  v8f sacc[4][4];
#pragma unroll
  for (int ti = 0; ti < 4; ++ti)
#pragma unroll
    for (int tj = 0; tj < 4; ++tj)
      sacc[ti][tj] = wmma16(qf[ti], kf[tj], (v8f){});

  // + relative position bias, softmax per row, write P (f16) to LDS.
#pragma unroll
  for (int ti = 0; ti < 4; ++ti) {
    float rmax[8], rsum[8];
#pragma unroll
    for (int tj = 0; tj < 4; ++tj) {
#pragma unroll
      for (int g = 0; g < 8; ++g) {
        int m = ti * 16 + hh * 8 + g;
        int n = tj * 16 + r;
        int di = (m >> 3) - (n >> 3) + 7;
        int dj = (m & 7) - (n & 7) + 7;
        sacc[ti][tj][g] += relb[(di * 15 + dj) * NHEADS + head];
      }
    }
#pragma unroll
    for (int g = 0; g < 8; ++g) {
      float mx = -3.0e38f;
#pragma unroll
      for (int tj = 0; tj < 4; ++tj) mx = fmaxf(mx, sacc[ti][tj][g]);
      rmax[g] = mx;
    }
#pragma unroll
    for (int off = 1; off < 16; off <<= 1)
#pragma unroll
      for (int g = 0; g < 8; ++g)
        rmax[g] = fmaxf(rmax[g], __shfl_xor(rmax[g], off, 32));
#pragma unroll
    for (int g = 0; g < 8; ++g) rsum[g] = 0.f;
#pragma unroll
    for (int tj = 0; tj < 4; ++tj) {
#pragma unroll
      for (int g = 0; g < 8; ++g) {
        float p = __expf(sacc[ti][tj][g] - rmax[g]);
        sacc[ti][tj][g] = p;
        rsum[g] += p;
      }
    }
#pragma unroll
    for (int off = 1; off < 16; off <<= 1)
#pragma unroll
      for (int g = 0; g < 8; ++g) rsum[g] += __shfl_xor(rsum[g], off, 32);
#pragma unroll
    for (int tj = 0; tj < 4; ++tj) {
#pragma unroll
      for (int g = 0; g < 8; ++g) {
        int m = ti * 16 + hh * 8 + g;
        int n = tj * 16 + r;
        P[wave][m * 64 + n] = (_Float16)(sacc[ti][tj][g] / rsum[g]);
      }
    }
  }
  __syncthreads();

  // V fragments (B operand): column = output dim d, K = key token.
  v16h vf[2][2];
#pragma unroll
  for (int kc = 0; kc < 2; ++kc) {
#pragma unroll
    for (int nj = 0; nj < 2; ++nj) {
      int d = nj * 16 + r;
#pragma unroll
      for (int e = 0; e < 16; ++e) {
        int key = kc * 32 + kpack(e, hh);
        vf[kc][nj][e] = rowbase[(size_t)key * (3 * CDIM) + 2 * CDIM + d];
      }
    }
  }

  // O = P @ V and store f16 to AttnOut[win*64+m, head*32+d].
#pragma unroll
  for (int ti = 0; ti < 4; ++ti) {
    v16h pa[2];
#pragma unroll
    for (int kc = 0; kc < 2; ++kc)
      pa[kc] = frag_ld(&P[wave][(ti * 16 + r) * 64 + kc * 32 + hh * 8]);
#pragma unroll
    for (int nj = 0; nj < 2; ++nj) {
      v8f oa = (v8f){};
#pragma unroll
      for (int kc = 0; kc < 2; ++kc) oa = wmma16(pa[kc], vf[kc][nj], oa);
#pragma unroll
      for (int g = 0; g < 8; ++g) {
        int m = ti * 16 + hh * 8 + g;
        int d = nj * 16 + r;
        attn_out[(size_t)(win * 64 + m) * CDIM + head * HDIM + d] =
            (_Float16)oa[g];
      }
    }
  }
}

// ---------------------------------------------------------------------------
// Gated SiLU:  G[m,j] = fc1[m,j] * silu(fc1[m,1536+j])
// ---------------------------------------------------------------------------
__global__ void silu_gate_kernel(const _Float16* __restrict__ fc1,
                                 _Float16* __restrict__ G) {
  int i = blockIdx.x * blockDim.x + threadIdx.x;
  int m = i / 1536, j = i % 1536;
  float v = (float)fc1[(size_t)m * 3072 + j];
  float g = (float)fc1[(size_t)m * 3072 + 1536 + j];
  float s = g / (1.f + __expf(-g));
  G[(size_t)m * 1536 + j] = (_Float16)(v * s);
}

// ---------------------------------------------------------------------------
// Workspace layout (bytes, all 16B-aligned)
// ---------------------------------------------------------------------------
constexpr size_t OFF_MOD1 = 0;
constexpr size_t OFF_MOD2 = OFF_MOD1 + 8 * 768 * 4;
constexpr size_t OFF_WQKV = OFF_MOD2 + 8 * 768 * 4;
constexpr size_t OFF_WPRJ = OFF_WQKV + (size_t)384 * 1152 * 2;
constexpr size_t OFF_WFC1 = OFF_WPRJ + (size_t)384 * 384 * 2;
constexpr size_t OFF_WFC2 = OFF_WFC1 + (size_t)384 * 3072 * 2;
constexpr size_t OFF_XWIN = OFF_WFC2 + (size_t)1536 * 384 * 2;
constexpr size_t OFF_ATTO = OFF_XWIN + (size_t)NTOK * CDIM * 2;
constexpr size_t OFF_QKV  = OFF_ATTO + (size_t)NTOK * CDIM * 2;
constexpr size_t OFF_H1   = OFF_QKV  + (size_t)NTOK * 1152 * 2;
constexpr size_t OFF_X2   = OFF_H1   + (size_t)NTOK * CDIM * 4;
constexpr size_t OFF_FC1  = OFF_X2   + (size_t)NTOK * CDIM * 2;
constexpr size_t OFF_G    = OFF_FC1  + (size_t)NTOK * 3072 * 2;

extern "C" void kernel_launch(void* const* d_in, const int* in_sizes, int n_in,
                              void* d_out, int out_size, void* d_ws, size_t ws_size,
                              hipStream_t stream) {
  const float* x       = (const float*)d_in[0];
  const float* t_emb   = (const float*)d_in[1];
  const float* qkv_w   = (const float*)d_in[2];
  const float* q_norm  = (const float*)d_in[3];
  const float* k_norm  = (const float*)d_in[4];
  const float* relb    = (const float*)d_in[5];
  const float* proj_w  = (const float*)d_in[6];
  const float* proj_b  = (const float*)d_in[7];
  const float* n1w     = (const float*)d_in[8];
  const float* n2w     = (const float*)d_in[9];
  const float* fc1_w   = (const float*)d_in[10];
  const float* fc1_b   = (const float*)d_in[11];
  const float* fc2_w   = (const float*)d_in[12];
  const float* fc2_b   = (const float*)d_in[13];
  const float* tp1_w   = (const float*)d_in[14];
  const float* tp1_b   = (const float*)d_in[15];
  const float* tp2_w   = (const float*)d_in[16];
  const float* tp2_b   = (const float*)d_in[17];

  char* ws = (char*)d_ws;
  float*    mod1 = (float*)(ws + OFF_MOD1);
  float*    mod2 = (float*)(ws + OFF_MOD2);
  _Float16* Wqkv = (_Float16*)(ws + OFF_WQKV);
  _Float16* Wprj = (_Float16*)(ws + OFF_WPRJ);
  _Float16* Wfc1 = (_Float16*)(ws + OFF_WFC1);
  _Float16* Wfc2 = (_Float16*)(ws + OFF_WFC2);
  _Float16* Xwin = (_Float16*)(ws + OFF_XWIN);
  _Float16* Atto = (_Float16*)(ws + OFF_ATTO);
  _Float16* QKV  = (_Float16*)(ws + OFF_QKV);
  float*    H1   = (float*)(ws + OFF_H1);
  _Float16* X2   = (_Float16*)(ws + OFF_X2);
  _Float16* FC1  = (_Float16*)(ws + OFF_FC1);
  _Float16* G    = (_Float16*)(ws + OFF_G);

  // Stage 0: time modulation tables.
  time_mod_kernel<<<(2 * 8 * 768 + 255) / 256, 256, 0, stream>>>(
      t_emb, tp1_w, tp1_b, tp2_w, tp2_b, mod1, mod2);

  // Stage 1: weight conversion to f16.
  cvt_f16_kernel<<<(384 * 1152 + 255) / 256, 256, 0, stream>>>(qkv_w, Wqkv, 384 * 1152);
  cvt_f16_kernel<<<(384 * 384 + 255) / 256, 256, 0, stream>>>(proj_w, Wprj, 384 * 384);
  cvt_f16_kernel<<<(384 * 3072 + 255) / 256, 256, 0, stream>>>(fc1_w, Wfc1, 384 * 3072);
  cvt_f16_kernel<<<(1536 * 384 + 255) / 256, 256, 0, stream>>>(fc2_w, Wfc2, 1536 * 384);

  // Stage 2: norm1 + shift + window partition -> Xwin (f16, windowed order).
  norm1_window_kernel<<<NTOK / 8, 256, 0, stream>>>(x, mod1, n1w, Xwin);

  // Stage 3: QKV GEMM  [32768,384] @ [384,1152] -> QKV f16.
  gemm_kernel<0><<<dim3(1152 / 64, NTOK / 128), 256, 0, stream>>>(
      Xwin, Wqkv, NTOK, 1152, 384, QKV, nullptr, nullptr, nullptr, nullptr);

  // Stage 4: window attention (512 windows x 12 heads) -> Atto f16.
  attn_kernel<<<(512 * NHEADS) / 4, 128, 0, stream>>>(QKV, q_norm, k_norm, relb, Atto);

  // Stage 5: proj GEMM + un-shift + residual + scale -> H1 (NHWC f32).
  gemm_kernel<1><<<dim3(384 / 64, NTOK / 128), 256, 0, stream>>>(
      Atto, Wprj, NTOK, 384, 384, nullptr, proj_b, x, H1, nullptr);

  // Stage 6: norm2 -> X2 f16.
  norm2_kernel<<<NTOK / 8, 256, 0, stream>>>(H1, mod2, n2w, X2);

  // Stage 7: fc1 GEMM [32768,384] @ [384,3072] -> FC1 f16 (with bias).
  gemm_kernel<2><<<dim3(3072 / 64, NTOK / 128), 256, 0, stream>>>(
      X2, Wfc1, NTOK, 3072, 384, FC1, fc1_b, nullptr, nullptr, nullptr);

  // Stage 8: SiLU gating -> G f16.
  silu_gate_kernel<<<(NTOK * 1536) / 256, 256, 0, stream>>>(FC1, G);

  // Stage 9: fc2 GEMM [32768,1536] @ [1536,384] + residual + NCHW store.
  gemm_kernel<3><<<dim3(384 / 64, NTOK / 128), 256, 0, stream>>>(
      G, Wfc2, NTOK, 384, 1536, nullptr, fc2_b, nullptr, H1, (float*)d_out);
}
